// _ObjectAttentionBlock_11020886081660
// MI455X (gfx1250) — matrix operand order repaired and checked by reading, not compiled, not run on hardware
//
#include <hip/hip_runtime.h>
#include <cstdint>

typedef __attribute__((ext_vector_type(16))) _Float16 v16h;
typedef __attribute__((ext_vector_type(8)))  _Float16 v8h;
typedef __attribute__((ext_vector_type(8)))  float    v8f;

#define SLOPE 0.01f
#define EPSIN 1e-5f

// LDS strides (halfs); all multiples of 16 halfs (32B) so v16h fragment loads are aligned.
#define AS_STRIDE 48
#define BS_STRIDE 48
#define QT_STRIDE 272
#define KH_STRIDE 272
#define VH_STRIDE 64
#define AT_STRIDE 64
#define CT_STRIDE 272

__device__ __forceinline__ float leaky(float v) { return v >= 0.f ? v : SLOPE * v; }

// K layout for A-operands: identity (no swizzle), so a lane's A fragment
// (K = 16h+0..15 with h = lane>=16) matches the contiguous B fragment order.
__device__ __forceinline__ int perm32(int k) {
  int r = k & 31;
  int p = r;
  return (k & ~31) | p;
}

__device__ __forceinline__ v8f wmma16(v16h a, v16h b, v8f c) {
  return __builtin_amdgcn_wmma_f32_16x16x32_f16(false, a, false, b, (short)0, c, false, false);
}

// ---------------------------------------------------------------- prep kernels

// f32 -> f16 with A-operand K-swizzle (row-major [rows][K], K multiple of 32)
__global__ void cvt_swz_f16(const float* __restrict__ s, _Float16* __restrict__ d,
                            int n, int K) {
  int i = blockIdx.x * blockDim.x + threadIdx.x;
  if (i >= n) return;
  int row = i / K, k = i - row * K;
  d[row * K + perm32(k)] = (_Float16)s[i];
}

// K = leaky(w_obj2 @ leaky(w_obj1 @ proxy)) stored TRANSPOSED f16 [N][64 m][256 kc]
// V = leaky(w_down @ proxy) stored f16 [N][256 kc][64 m] with m swizzled (A-operand of ctx GEMM)
__global__ __launch_bounds__(256) void kv_kernel(const float* __restrict__ proxy,
                                                 const float* __restrict__ w1,
                                                 const float* __restrict__ w2,
                                                 const float* __restrict__ wd,
                                                 _Float16* __restrict__ KhT,
                                                 _Float16* __restrict__ Vh) {
  int m = blockIdx.x, n = blockIdx.y, t = threadIdx.x;
  __shared__ float psm[512];
  __shared__ float k1s[256];
  psm[t]       = proxy[((size_t)n * 512 + t) * 64 + m];
  psm[t + 256] = proxy[((size_t)n * 512 + t + 256) * 64 + m];
  __syncthreads();
  float a1 = 0.f, av = 0.f;
  const float* r1 = w1 + (size_t)t * 512;
  const float* rd = wd + (size_t)t * 512;
  for (int c = 0; c < 512; ++c) { float p = psm[c]; a1 += r1[c] * p; av += rd[c] * p; }
  float k1 = leaky(a1);
  float v  = leaky(av);
  k1s[t] = k1;
  __syncthreads();
  float a2 = 0.f;
  const float* r2 = w2 + (size_t)t * 256;
  for (int c = 0; c < 256; ++c) a2 += r2[c] * k1s[c];
  KhT[((size_t)n * 64 + m) * 256 + t] = (_Float16)leaky(a2);
  Vh[((size_t)n * 256 + t) * 64 + perm32(m)] = (_Float16)v;
}

// mean / rsqrt(var+eps) over one row of length S (biased var, matches jnp.var)
__global__ __launch_bounds__(256) void row_stats(const float* __restrict__ Y,
                                                 float* __restrict__ mean,
                                                 float* __restrict__ rstd, int S) {
  int row = blockIdx.x;
  const float4* p4 = (const float4*)(Y + (size_t)row * S);
  int n4 = S >> 2;
  float s = 0.f, s2 = 0.f;
  for (int i = threadIdx.x; i < n4; i += 256) {
    float4 v = p4[i];
    s  += v.x + v.y + v.z + v.w;
    s2 += v.x * v.x + v.y * v.y + v.z * v.z + v.w * v.w;
  }
  __shared__ float rs[256], rq[256];
  rs[threadIdx.x] = s; rq[threadIdx.x] = s2;
  __syncthreads();
  for (int o = 128; o > 0; o >>= 1) {
    if (threadIdx.x < o) { rs[threadIdx.x] += rs[threadIdx.x + o]; rq[threadIdx.x] += rq[threadIdx.x + o]; }
    __syncthreads();
  }
  if (threadIdx.x == 0) {
    float m = rs[0] / (float)S;
    float var = rq[0] / (float)S - m * m;
    mean[row] = m;
    rstd[row] = rsqrtf(var + EPSIN);
  }
}

// ---------------------------------------------------------------- main GEMM
// Y[n][M][S] = A_f16[M][K](K-swizzled) @ f(B[n][K][S]).
// MODE 0: f = cvt; MODE 1: f = leaky((b-mean)*rstd) fused InstanceNorm+LeakyReLU.
// Block tile 64x128, 8 waves: each wave = 1 row-tile x 4 col-tiles, K-step 32.
// B staged TRANSPOSED in LDS (float4 global loads) so fragments are contiguous v16h loads.
template <int MODE>
__global__ __launch_bounds__(256) void gemm_f16(const _Float16* __restrict__ A,
                                                const float* __restrict__ B,
                                                float* __restrict__ Y,
                                                const float* __restrict__ mean,
                                                const float* __restrict__ rstd,
                                                int Mrows, int K, int S) {
  __shared__ _Float16 As[64 * AS_STRIDE];
  __shared__ _Float16 BsT[128 * BS_STRIDE];
  int n = blockIdx.z;
  const float* Bp = B + (size_t)n * K * S;
  float* Yp = Y + (size_t)n * Mrows * S;
  const float* mp = mean + n * K;
  const float* rp = rstd + n * K;
  int s0 = blockIdx.x * 128, m0 = blockIdx.y * 64;
  int t = threadIdx.x, lane = t & 31, wv = t >> 5;
  int tr = wv >> 1, tc0 = (wv & 1) * 4;
  int hh = (lane >> 4) << 4;
  v8f acc[4] = {};
  for (int k0 = 0; k0 < K; k0 += 32) {
    { // A tile 64x32 (already K-swizzled in global): contiguous 16B copies
      int r = t >> 2, seg = (t & 3) << 3;
      *(v8h*)&As[r * AS_STRIDE + seg] = *(const v8h*)(A + (size_t)(m0 + r) * K + k0 + seg);
    }
#pragma unroll
    for (int i = 0; i < 4; ++i) { // B tile 32x128: float4 loads -> transposed [col][k] f16
      int lin = t + (i << 8);           // chunk id, 1024 chunks of 4 floats
      int kr = lin >> 5;                // 32 chunks per K-row
      int sc = (lin & 31) << 2;
      float4 vv = *(const float4*)(Bp + (size_t)(k0 + kr) * S + s0 + sc);
      if (MODE) {
        float mu = mp[k0 + kr], rsd = rp[k0 + kr];
        vv.x = leaky((vv.x - mu) * rsd);
        vv.y = leaky((vv.y - mu) * rsd);
        vv.z = leaky((vv.z - mu) * rsd);
        vv.w = leaky((vv.w - mu) * rsd);
      }
      BsT[(sc + 0) * BS_STRIDE + kr] = (_Float16)vv.x;
      BsT[(sc + 1) * BS_STRIDE + kr] = (_Float16)vv.y;
      BsT[(sc + 2) * BS_STRIDE + kr] = (_Float16)vv.z;
      BsT[(sc + 3) * BS_STRIDE + kr] = (_Float16)vv.w;
    }
    __syncthreads();
    v16h a = *(const v16h*)&As[(tr * 16 + (lane & 15)) * AS_STRIDE + hh];
#pragma unroll
    for (int c = 0; c < 4; ++c) {
      v16h b = *(const v16h*)&BsT[((tc0 + c) * 16 + (lane & 15)) * BS_STRIDE + hh];
      acc[c] = wmma16(a, b, acc[c]);
    }
    __syncthreads();
  }
  int rbase = m0 + tr * 16 + ((lane >> 4) << 3);
#pragma unroll
  for (int c = 0; c < 4; ++c) {
    int col = s0 + (tc0 + c) * 16 + (lane & 15);
#pragma unroll
    for (int j = 0; j < 8; ++j) Yp[(size_t)(rbase + j) * S + col] = acc[c][j];
  }
}

// ---------------------------------------------------------------- fused attention
// Per block: 32 spatial columns. q = leaky(norm(y2)); sim = q^T K / 16; softmax(M=64);
// ctx = V attn^T; y3 = w_up ctx.  63KB LDS, phase-overlapped regions.
__global__ __launch_bounds__(256) void attn_kernel(const float* __restrict__ y2,
                                                   const _Float16* __restrict__ KTg,
                                                   const _Float16* __restrict__ Vg,
                                                   const _Float16* __restrict__ wuh,
                                                   const float* __restrict__ mean2,
                                                   const float* __restrict__ rstd2,
                                                   float* __restrict__ y3, int S) {
  __shared__ __align__(32) char smem[64512];
  _Float16* KV    = (_Float16*)smem;             // KhT [64][KH_STRIDE] then Vh [256][VH_STRIDE]
  _Float16* qT    = (_Float16*)(smem + 34816);   // qT [32][QT_STRIDE], reused as ctxT [32][CT_STRIDE]
  float*    simS  = (float*)(smem + 52224);      // [32][64] f32
  _Float16* attnS = (_Float16*)(smem + 60416);   // [32][AT_STRIDE] f16

  int n = blockIdx.y, s0 = blockIdx.x * 32;
  int t = threadIdx.x, lane = t & 31, wv = t >> 5;
  int hh = (lane >> 4) << 4;

  { // K^T tile [64 m][256 kc] -> LDS rows of KH_STRIDE
    const unsigned int* s = (const unsigned int*)(KTg + (size_t)n * 64 * 256);
    unsigned int* d = (unsigned int*)KV;
#pragma unroll
    for (int i = 0; i < 32; ++i) {
      int idx = t + (i << 8);
      int r = idx >> 7, c = idx & 127;
      d[r * (KH_STRIDE / 2) + c] = s[idx];
    }
  }
  { // q^T [32 s][256 kc]: float4 loads, fused InstanceNorm+LeakyReLU, K-swizzled store
    const float* m2 = mean2 + n * 256;
    const float* r2 = rstd2 + n * 256;
#pragma unroll
    for (int i = 0; i < 8; ++i) {
      int lin = t + (i << 8);           // 2048 chunks of 4 floats
      int kc = lin >> 3;                // 8 chunks per kc-row of 32
      int c4 = (lin & 7) << 2;
      float4 vv = *(const float4*)(y2 + ((size_t)(n * 256 + kc)) * S + s0 + c4);
      float mu = m2[kc], rsd = r2[kc];
      int pk = perm32(kc);
      qT[(c4 + 0) * QT_STRIDE + pk] = (_Float16)leaky((vv.x - mu) * rsd);
      qT[(c4 + 1) * QT_STRIDE + pk] = (_Float16)leaky((vv.y - mu) * rsd);
      qT[(c4 + 2) * QT_STRIDE + pk] = (_Float16)leaky((vv.z - mu) * rsd);
      qT[(c4 + 3) * QT_STRIDE + pk] = (_Float16)leaky((vv.w - mu) * rsd);
    }
  }
  __syncthreads();

  { // sim = q^T K : [32 s][64 m], one 16x16 tile per wave, KC reduce in 8 steps
    int tr = wv >> 2, tc = wv & 3;
    v8f acc = {};
#pragma unroll
    for (int ks = 0; ks < 8; ++ks) {
      v16h a = *(const v16h*)&qT[(tr * 16 + (lane & 15)) * QT_STRIDE + ks * 32 + hh];
      v16h b = *(const v16h*)&KV[(tc * 16 + (lane & 15)) * KH_STRIDE + ks * 32 + hh];
      acc = wmma16(a, b, acc);
    }
    int srow = tr * 16 + ((lane >> 4) << 3);
    int mcol = tc * 16 + (lane & 15);
#pragma unroll
    for (int j = 0; j < 8; ++j) simS[(srow + j) * 64 + mcol] = acc[j] * 0.0625f; // KC^-0.5
  }
  __syncthreads();

  { // overwrite K region with V [256 kc][64 m] (m pre-swizzled in global)
    const unsigned int* s = (const unsigned int*)(Vg + (size_t)n * 256 * 64);
    unsigned int* d = (unsigned int*)KV;
#pragma unroll
    for (int i = 0; i < 32; ++i) d[t + (i << 8)] = s[t + (i << 8)];
  }
  if (t < 32) { // softmax over M=64; write attn [32 s][64 m] f16 (B-operand: [col][K])
    float mx = -1e30f;
    for (int m = 0; m < 64; ++m) mx = fmaxf(mx, simS[t * 64 + m]);
    float sum = 0.f;
    for (int m = 0; m < 64; ++m) { float e = __expf(simS[t * 64 + m] - mx); simS[t * 64 + m] = e; sum += e; }
    float inv = 1.f / sum;
    for (int m = 0; m < 64; ++m) attnS[t * AT_STRIDE + m] = (_Float16)(simS[t * 64 + m] * inv);
  }
  __syncthreads();

  _Float16* ctxT = qT; // reuse region (q fully consumed)
  { // ctx = V attn^T : [256 kc][32 s]; wave owns 2 row-tiles x 2 col-tiles, K=64 in 2 steps
    v8f acc[2][2] = {};
#pragma unroll
    for (int ks = 0; ks < 2; ++ks) {
      v16h b0 = *(const v16h*)&attnS[(lane & 15) * AT_STRIDE + ks * 32 + hh];
      v16h b1 = *(const v16h*)&attnS[(16 + (lane & 15)) * AT_STRIDE + ks * 32 + hh];
#pragma unroll
      for (int rt = 0; rt < 2; ++rt) {
        v16h a = *(const v16h*)&KV[((wv * 2 + rt) * 16 + (lane & 15)) * VH_STRIDE + ks * 32 + hh];
        acc[rt][0] = wmma16(a, b0, acc[rt][0]);
        acc[rt][1] = wmma16(a, b1, acc[rt][1]);
      }
    }
#pragma unroll
    for (int rt = 0; rt < 2; ++rt) {
      int row = (wv * 2 + rt) * 16 + ((lane >> 4) << 3);
#pragma unroll
      for (int ct = 0; ct < 2; ++ct) {
        int col = ct * 16 + (lane & 15);
#pragma unroll
        for (int j = 0; j < 8; ++j) ctxT[col * CT_STRIDE + row + j] = (_Float16)acc[rt][ct][j];
      }
    }
  }
  __syncthreads();

  { // y3 = w_up ctx : [512 c][32 s]; A streamed from global f16 (K-swizzled, L2 resident)
    v8f acc[4][2] = {};
#pragma unroll
    for (int ks = 0; ks < 8; ++ks) {
      v16h b0 = *(const v16h*)&ctxT[(lane & 15) * CT_STRIDE + ks * 32 + hh];
      v16h b1 = *(const v16h*)&ctxT[(16 + (lane & 15)) * CT_STRIDE + ks * 32 + hh];
#pragma unroll
      for (int rt = 0; rt < 4; ++rt) {
        v16h a = *(const v16h*)(wuh + (size_t)((wv * 4 + rt) * 16 + (lane & 15)) * 256 + ks * 32 + hh);
        acc[rt][0] = wmma16(a, b0, acc[rt][0]);
        acc[rt][1] = wmma16(a, b1, acc[rt][1]);
      }
    }
    float* y3p = y3 + (size_t)n * 512 * S;
#pragma unroll
    for (int rt = 0; rt < 4; ++rt) {
      int row = (wv * 4 + rt) * 16 + ((lane >> 4) << 3);
#pragma unroll
      for (int ct = 0; ct < 2; ++ct) {
        int col = s0 + ct * 16 + (lane & 15);
#pragma unroll
        for (int j = 0; j < 8; ++j) y3p[(size_t)(row + j) * S + col] = acc[rt][ct][j];
      }
    }
  }
}

// out = leaky(norm(y3)) elementwise, float4 vectorized
__global__ __launch_bounds__(256) void finalize_kernel(const float* __restrict__ y3,
                                                       const float* __restrict__ mean,
                                                       const float* __restrict__ rstd,
                                                       float* __restrict__ out, int S) {
  int row = blockIdx.z * gridDim.y + blockIdx.y;
  int c4 = (blockIdx.x * 256 + threadIdx.x) << 2;
  size_t idx = (size_t)row * S + c4;
  float4 v = *(const float4*)(y3 + idx);
  float mu = mean[row], rs = rstd[row];
  v.x = leaky((v.x - mu) * rs);
  v.y = leaky((v.y - mu) * rs);
  v.z = leaky((v.z - mu) * rs);
  v.w = leaky((v.w - mu) * rs);
  *(float4*)(out + idx) = v;
}

// ---------------------------------------------------------------- launch

extern "C" void kernel_launch(void* const* d_in, const int* in_sizes, int n_in,
                              void* d_out, int out_size, void* d_ws, size_t ws_size,
                              hipStream_t stream) {
  (void)in_sizes; (void)n_in; (void)out_size; (void)ws_size;
  const float* x      = (const float*)d_in[0];
  const float* proxy  = (const float*)d_in[1];
  const float* w_pix1 = (const float*)d_in[2];
  const float* w_pix2 = (const float*)d_in[3];
  const float* w_obj1 = (const float*)d_in[4];
  const float* w_obj2 = (const float*)d_in[5];
  const float* w_down = (const float*)d_in[6];
  const float* w_up   = (const float*)d_in[7];
  float* out = (float*)d_out;

  const int Nb = 2, C = 512, KC = 256, M = 64, S = 32 * 32 * 32;

  char* w = (char*)d_ws;
  size_t off = 0;
  auto take = [&](size_t bytes) { char* p = w + off; off = (off + bytes + 255) & ~(size_t)255; return p; };

  _Float16* w1h  = (_Float16*)take((size_t)KC * C * 2);
  _Float16* w2h  = (_Float16*)take((size_t)KC * KC * 2);
  _Float16* wuh  = (_Float16*)take((size_t)C * KC * 2);
  _Float16* KhTg = (_Float16*)take((size_t)Nb * KC * M * 2);
  _Float16* Vhg  = (_Float16*)take((size_t)Nb * KC * M * 2);
  float* mean1 = (float*)take((size_t)Nb * KC * 4);
  float* rstd1 = (float*)take((size_t)Nb * KC * 4);
  float* mean2 = (float*)take((size_t)Nb * KC * 4);
  float* rstd2 = (float*)take((size_t)Nb * KC * 4);
  float* mean3 = (float*)take((size_t)Nb * C * 4);
  float* rstd3 = (float*)take((size_t)Nb * C * 4);
  // big buffers: yA holds y1 (first half) then is overwritten by y3 (y1 dead after gemm2)
  float* yA = (float*)take((size_t)Nb * C * S * 4);    // 134 MB
  float* y2 = (float*)take((size_t)Nb * KC * S * 4);   //  67 MB
  float* y1 = yA;
  float* y3 = yA;

  cvt_swz_f16<<<dim3((KC * C + 255) / 256), 256, 0, stream>>>(w_pix1, w1h, KC * C, C);
  cvt_swz_f16<<<dim3((KC * KC + 255) / 256), 256, 0, stream>>>(w_pix2, w2h, KC * KC, KC);
  cvt_swz_f16<<<dim3((C * KC + 255) / 256), 256, 0, stream>>>(w_up, wuh, C * KC, KC);
  kv_kernel<<<dim3(M, Nb), 256, 0, stream>>>(proxy, w_obj1, w_obj2, w_down, KhTg, Vhg);

  gemm_f16<0><<<dim3(S / 128, KC / 64, Nb), 256, 0, stream>>>(w1h, x, y1, mean1, rstd1, KC, C, S);
  row_stats<<<dim3(Nb * KC), 256, 0, stream>>>(y1, mean1, rstd1, S);
  gemm_f16<1><<<dim3(S / 128, KC / 64, Nb), 256, 0, stream>>>(w2h, y1, y2, mean1, rstd1, KC, KC, S);
  row_stats<<<dim3(Nb * KC), 256, 0, stream>>>(y2, mean2, rstd2, S);

  attn_kernel<<<dim3(S / 32, Nb), 256, 0, stream>>>(y2, KhTg, Vhg, wuh, mean2, rstd2, y3, S);

  row_stats<<<dim3(Nb * C), 256, 0, stream>>>(y3, mean3, rstd3, S);
  finalize_kernel<<<dim3(S / 1024, C, Nb), 256, 0, stream>>>(y3, mean3, rstd3, out, S);
}
